// CTNNBackflowNet_18056042512511
// MI455X (gfx1250) — matrix operand order repaired and checked by reading, not compiled, not run on hardware
//
#include <hip/hip_runtime.h>
#include <hip/hip_bf16.h>
#include <math.h>

// ---------------------------------------------------------------------------
// CDNA5 (gfx1250) fused implementation of the backflow message-passing net.
// Heavy edge GEMMs use v_wmma_f32_16x16x32_f16 (f16 in, f32 accum).
// Weights are pre-swizzled into B-fragment order so every fragment is two
// ds_load_b128 per lane; A fragments are register-cached across column tiles.
// Weight staging uses the gfx1250 async global->LDS engine (ASYNCcnt).
// ---------------------------------------------------------------------------

typedef _Float16 v16h __attribute__((ext_vector_type(16)));
typedef _Float16 v8h  __attribute__((ext_vector_type(8)));
typedef _Float16 v2h  __attribute__((ext_vector_type(2)));
typedef float    v8f  __attribute__((ext_vector_type(8)));
typedef int      bf_v4i __attribute__((ext_vector_type(4)));
typedef __attribute__((address_space(1))) bf_v4i* bf_gp;   // global int4*
typedef __attribute__((address_space(3))) bf_v4i* bf_lp;   // LDS int4*

#define BATCH 16
#define NN    128
#define HH    128
#define SA    136        // activation LDS stride in halfs (row step 272B, 16B-aligned)
#define LANE_H 24        // swizzled-weight per-lane stride in halfs (48B, 16B-aligned)
#define TILE_H (32 * LANE_H)          // 768 halfs per 32x16 fragment tile
#define WSWZ_H (32 * TILE_H)          // 24576 halfs per 128x128 matrix

#if defined(__gfx1250__) && __has_builtin(__builtin_amdgcn_global_load_async_to_lds_b128) && __has_builtin(__builtin_amdgcn_s_wait_asynccnt)
#define BF_ASYNC_LDS 1
#else
#define BF_ASYNC_LDS 0
#endif

__device__ __forceinline__ float silu_f(float v) {
    return v / (1.0f + __expf(-v));
}

// Swizzled position of W[k][col] (K=128 x N=128) in B-fragment order:
// tile = (ct*4 + ks); within tile lane L = 16*((kr>>3)&1) + n gets 16 halfs:
// k = 8h+s for s<8, k = 16+8h+(s-8) for s>=8 (matches A fragment order).
__device__ __forceinline__ int swz_pos(int k, int col) {
    const int ks = k >> 5;
    const int kr = k & 31;
    const int h  = (kr >> 3) & 1;
    const int vv = (kr >> 4) * 4 + ((kr & 7) >> 1);
    const int e  = kr & 1;
    const int ct = col >> 4;
    const int n  = col & 15;
    const int L  = h * 16 + n;
    return (ct * 4 + ks) * TILE_H + L * LANE_H + 2 * vv + e;
}

// Load the 4 A fragments (K=0..127) of one 16-row tile into registers.
// Per lane: rows m = lane&15; halfs [h*8 .. h*8+7] and [16+h*8 .. 16+h*8+7].
__device__ __forceinline__ void load_afrags(const _Float16* __restrict__ Ain,
                                            int lane, v16h (&a)[4]) {
    const int m = lane & 15;
    const int h = lane >> 4;
    const _Float16* ap = Ain + m * SA + h * 8;
    #pragma unroll
    for (int ks = 0; ks < 4; ++ks) {
        v8h lo = *(const v8h*)(ap + ks * 32);
        v8h hi = *(const v8h*)(ap + ks * 32 + 16);
        a[ks] = __builtin_shufflevector(lo, hi, 0, 1, 2, 3, 4, 5, 6, 7,
                                        8, 9, 10, 11, 12, 13, 14, 15);
    }
}

// One 16x16 output tile: 4 k-steps against the swizzled weight in LDS.
__device__ __forceinline__ v8f wmma_row(const v16h (&a)[4],
                                        const _Float16* __restrict__ Wswz,
                                        int ct, int lane) {
    v8f acc = {0.f, 0.f, 0.f, 0.f, 0.f, 0.f, 0.f, 0.f};
    const _Float16* bp = Wswz + (ct * 4) * TILE_H + lane * LANE_H;
    #pragma unroll
    for (int ks = 0; ks < 4; ++ks) {
        v8h lo = *(const v8h*)(bp + ks * TILE_H);
        v8h hi = *(const v8h*)(bp + ks * TILE_H + 8);
        v16h bf = __builtin_shufflevector(lo, hi, 0, 1, 2, 3, 4, 5, 6, 7,
                                          8, 9, 10, 11, 12, 13, 14, 15);
        acc = __builtin_amdgcn_wmma_f32_16x16x32_f16(
            /*neg_a=*/false, a[ks], /*neg_b=*/false, bf,
            /*c_mod=*/(short)0, acc, /*reuse_a=*/false, /*reuse_b=*/false);
    }
    return acc;
}

// ---------------------------------------------------------------------------
// Kernel 0: weight prep -> swizzled f16 fragment order in global memory.
// wE = ee2_w, wU = eu1_w[0:128] (edge part), wM = eu2_w @ rev_w (fused
// transport), bmsg = eu2_b @ rev_w.  grid 128 x 128: block=k row, thread=col.
// ---------------------------------------------------------------------------
__global__ void bf_prep_kernel(const float* __restrict__ ee2_w,
                               const float* __restrict__ eu1_w,
                               const float* __restrict__ eu2_w,
                               const float* __restrict__ rev_w,
                               const float* __restrict__ eu2_b,
                               _Float16* __restrict__ wE,
                               _Float16* __restrict__ wU,
                               _Float16* __restrict__ wM,
                               float* __restrict__ bmsg) {
    const int r = blockIdx.x;
    const int c = threadIdx.x;
    const int p = swz_pos(r, c);
    wE[p] = (_Float16)ee2_w[r * HH + c];
    wU[p] = (_Float16)eu1_w[r * HH + c];
    float s = 0.f;
    for (int k = 0; k < HH; ++k)
        s += eu2_w[r * HH + k] * rev_w[k * HH + c];
    wM[p] = (_Float16)s;
    if (r == 0) {
        float t = 0.f;
        for (int k = 0; k < HH; ++k)
            t += eu2_b[k] * rev_w[k * HH + c];
        bmsg[c] = t;
    }
}

// ---------------------------------------------------------------------------
// Kernel 1: per-node precompute. h_v, vie = h_v@rve, hoisted edge_update node
// terms: pre_i = vie@eu1_w[128:256] + eu1_b, vjc = vie@eu1_w[256:384].
// ---------------------------------------------------------------------------
__global__ void bf_nodepre_kernel(const float* __restrict__ x,
                                  const int* __restrict__ spin,
                                  const float* __restrict__ ne_w,
                                  const float* __restrict__ ne_b,
                                  const float* __restrict__ rve_w,
                                  const float* __restrict__ eu1_w,
                                  const float* __restrict__ eu1_b,
                                  float* __restrict__ hv_g,
                                  float* __restrict__ prei_g,
                                  float* __restrict__ vjc_g) {
    __shared__ float hvrow[HH];
    __shared__ float vie[HH];
    const int row = blockIdx.x;
    const int f = threadIdx.x;
    const int n = row & (NN - 1);
    const float x0 = x[row * 3 + 0];
    const float x1 = x[row * 3 + 1];
    const float x2 = x[row * 3 + 2];
    const float sp = (float)spin[n];
    float h = x0 * ne_w[f] + x1 * ne_w[HH + f] + x2 * ne_w[2 * HH + f] +
              sp * ne_w[3 * HH + f] + ne_b[f];
    hv_g[row * HH + f] = h;
    hvrow[f] = h;
    __syncthreads();
    float v = 0.f;
    for (int k = 0; k < HH; ++k)
        v += hvrow[k] * rve_w[k * HH + f];
    vie[f] = v;
    __syncthreads();
    float pi = eu1_b[f];
    float vj = 0.f;
    for (int k = 0; k < HH; ++k) {
        const float vk = vie[k];
        pi += vk * eu1_w[(HH + k) * HH + f];
        vj += vk * eu1_w[(2 * HH + k) * HH + f];
    }
    prei_g[row * HH + f] = pi;
    vjc_g[row * HH + f] = vj;
}

// ---------------------------------------------------------------------------
// Kernel 2: fused edge pipeline. One workgroup = (batch b, 16-row i-tile,
// half the j range). 256 threads = 8 wave32. Per 16-j tile: edge features ->
// silu(ee1) scalar (K=5) -> WMMA ee2 -> WMMA eu1_edge (+node terms, silu)
// -> WMMA W_msg -> diag-masked register reduction into m_v partials.
// ---------------------------------------------------------------------------
__global__ __launch_bounds__(256) void bf_edge_kernel(
        const float* __restrict__ x,
        const _Float16* __restrict__ wE,
        const _Float16* __restrict__ wU,
        const _Float16* __restrict__ wM,
        const float* __restrict__ ee1_w,
        const float* __restrict__ ee1_b,
        const float* __restrict__ ee2_b,
        const float* __restrict__ bmsg,
        const float* __restrict__ prei_g,
        const float* __restrict__ vjc_g,
        float* __restrict__ mpart) {
    extern __shared__ char smem[];
    float* fbase = (float*)smem;
    float* sE1w  = fbase;            // 5*128
    float* sE1b  = sE1w + 640;       // 128
    float* sB2   = sE1b + 128;       // 128
    float* sBm   = sB2 + 128;        // 128
    float* sPreI = sBm + 128;        // 16*128
    float* sVjc  = sPreI + 2048;     // 16*128
    float* sXi   = sVjc + 2048;      // 16*3
    float* sXj   = sXi + 48;         // 16*3 (+pad to keep halves 16B-aligned)
    _Float16* hbase = (_Float16*)(sXj + 48);   // byte offset 20864 (16B-aligned)
    _Float16* sW2 = hbase;                     // WSWZ_H halfs each
    _Float16* sWu = sW2 + WSWZ_H;
    _Float16* sWm = sWu + WSWZ_H;
    _Float16* sA0 = sWm + WSWZ_H;              // 256*SA halfs
    _Float16* sA1 = sA0 + 256 * SA;

    const int tid   = threadIdx.x;
    const int lane  = tid & 31;
    const int wave  = tid >> 5;
    const int bid   = blockIdx.x;
    const int jhalf = bid & 1;
    const int itile = (bid >> 1) & 7;
    const int b     = bid >> 4;
    const int i0    = itile * 16;

    // Stage swizzled weights into LDS: async global->LDS engine when
    // available (ASYNCcnt-tracked, no VGPR round trip), else vector copies.
    {
        const uint4* gE = (const uint4*)wE;
        const uint4* gU = (const uint4*)wU;
        const uint4* gM = (const uint4*)wM;
        uint4* dE = (uint4*)sW2;
        uint4* dU = (uint4*)sWu;
        uint4* dM = (uint4*)sWm;
#if BF_ASYNC_LDS
        for (int idx = tid; idx < WSWZ_H / 8; idx += 256) {
            __builtin_amdgcn_global_load_async_to_lds_b128(
                (bf_gp)(gE + idx), (bf_lp)(dE + idx), 0, 0);
            __builtin_amdgcn_global_load_async_to_lds_b128(
                (bf_gp)(gU + idx), (bf_lp)(dU + idx), 0, 0);
            __builtin_amdgcn_global_load_async_to_lds_b128(
                (bf_gp)(gM + idx), (bf_lp)(dM + idx), 0, 0);
        }
        __builtin_amdgcn_s_wait_asynccnt(0);
#else
        for (int idx = tid; idx < WSWZ_H / 8; idx += 256) {
            dE[idx] = gE[idx];
            dU[idx] = gU[idx];
            dM[idx] = gM[idx];
        }
#endif
    }
    for (int idx = tid; idx < 640; idx += 256) sE1w[idx] = ee1_w[idx];
    if (tid < 128) {
        sE1b[tid] = ee1_b[tid];
        sB2[tid]  = ee2_b[tid];
        sBm[tid]  = bmsg[tid];
    }
    for (int idx = tid; idx < 2048; idx += 256)
        sPreI[idx] = prei_g[(b * NN + i0) * HH + idx];
    if (tid < 48) sXi[tid] = x[(b * NN + i0) * 3 + tid];

    float macc[16];
    #pragma unroll
    for (int t = 0; t < 16; ++t) macc[t] = 0.f;

    for (int jt = jhalf * 4; jt < jhalf * 4 + 4; ++jt) {
        const int j0 = jt * 16;
        __syncthreads();
        for (int idx = tid; idx < 2048; idx += 256)
            sVjc[idx] = vjc_g[(b * NN + j0) * HH + idx];
        if (tid < 48) sXj[tid] = x[(b * NN + j0) * 3 + tid];
        // Prefetch next j-tile's vjc rows into cache (global_prefetch_b8).
        if (jt + 1 < jhalf * 4 + 4) {
            const float* nxt = vjc_g + (b * NN + j0 + 16) * HH;
            __builtin_prefetch(nxt + tid * 8, 0, 0);
        }
        __syncthreads();

        // ---- GEMM1 (scalar, K=5): edge features -> silu(ee1) -> sA0 f16 ----
        {
            const int ii = tid >> 4, jj = tid & 15;
            const float r0 = sXi[ii * 3 + 0] - sXj[jj * 3 + 0];
            const float r1 = sXi[ii * 3 + 1] - sXj[jj * 3 + 1];
            const float r2 = sXi[ii * 3 + 2] - sXj[jj * 3 + 2];
            const float rsq = r0 * r0 + r1 * r1 + r2 * r2;
            const float rnm = sqrtf(rsq + 1e-12f);
            #pragma unroll 2
            for (int f = 0; f < 128; f += 2) {
                float va = r0 * sE1w[f] + r1 * sE1w[128 + f] + r2 * sE1w[256 + f] +
                           rnm * sE1w[384 + f] + rsq * sE1w[512 + f] + sE1b[f];
                float vb = r0 * sE1w[f + 1] + r1 * sE1w[129 + f] + r2 * sE1w[257 + f] +
                           rnm * sE1w[385 + f] + rsq * sE1w[513 + f] + sE1b[f + 1];
                v2h pk;
                pk[0] = (_Float16)silu_f(va);
                pk[1] = (_Float16)silu_f(vb);
                *(v2h*)(sA0 + tid * SA + f) = pk;   // ds_store_b32
            }
        }
        __syncthreads();

        // ---- Stage 2 (WMMA): h_e = sA0 @ ee2 + ee2_b -> sA1 (f16) ----
        #pragma unroll
        for (int rr = 0; rr < 2; ++rr) {
            const int rt = wave * 2 + rr;
            v16h afr[4];
            load_afrags(sA0 + rt * 16 * SA, lane, afr);
            for (int ct = 0; ct < 8; ++ct) {
                v8f acc = wmma_row(afr, sW2, ct, lane);
                const int n = lane & 15, h = lane >> 4;
                const int col = ct * 16 + n;
                const float bias = sB2[col];
                #pragma unroll
                for (int v = 0; v < 8; ++v) {
                    const int m = v + 8 * h;
                    sA1[(rt * 16 + m) * SA + col] = (_Float16)(acc[v] + bias);
                }
            }
        }
        __syncthreads();

        // ---- Stage 3 (WMMA): t = silu(h_e @ eu1_lo + pre_i + vjc) -> sA0 ----
        #pragma unroll
        for (int rr = 0; rr < 2; ++rr) {
            const int rt = wave * 2 + rr;
            v16h afr[4];
            load_afrags(sA1 + rt * 16 * SA, lane, afr);
            for (int ct = 0; ct < 8; ++ct) {
                v8f acc = wmma_row(afr, sWu, ct, lane);
                const int n = lane & 15, h = lane >> 4;
                const int col = ct * 16 + n;
                const float pi = sPreI[rt * HH + col];
                #pragma unroll
                for (int v = 0; v < 8; ++v) {
                    const int m = v + 8 * h;      // tile row == local j index
                    const float pre = acc[v] + pi + sVjc[m * HH + col];
                    sA0[(rt * 16 + m) * SA + col] = (_Float16)silu_f(pre);
                }
            }
        }
        __syncthreads();

        // ---- Stage 4 (WMMA): msg = t @ W_msg + b_msg, diag-mask, reduce ----
        #pragma unroll
        for (int rr = 0; rr < 2; ++rr) {
            const int rt = wave * 2 + rr;
            v16h afr[4];
            load_afrags(sA0 + rt * 16 * SA, lane, afr);
            for (int ct = 0; ct < 8; ++ct) {
                v8f acc = wmma_row(afr, sWm, ct, lane);
                const int n = lane & 15, h = lane >> 4;
                const int col = ct * 16 + n;
                const float bm = sBm[col];
                const int ig = i0 + rt;
                float s = 0.f;
                #pragma unroll
                for (int v = 0; v < 8; ++v) {
                    const int m = v + 8 * h;
                    const int jg = j0 + m;
                    if (jg != ig) s += acc[v] + bm;   // (1 - eye) mask
                }
                macc[rr * 8 + ct] += s;
            }
        }
        __syncthreads();
    }

    // Combine lane halves (rows 0..7 vs 8..15) and emit m_v partials.
    for (int t = 0; t < 16; ++t) {
        const int rt = wave * 2 + (t >> 3);
        const int ct = t & 7;
        const float tot = macc[t] + __shfl_xor(macc[t], 16, 32);
        if (lane < 16) {
            const size_t idx =
                (((size_t)jhalf * BATCH + b) * NN + (i0 + rt)) * HH + ct * 16 + lane;
            mpart[idx] = tot;
        }
    }
}

// ---------------------------------------------------------------------------
// Kernel 3: node_update MLP + dx head (pre-mean).  grid B*N x 128 threads.
// ---------------------------------------------------------------------------
__global__ void bf_nodeupd_kernel(const float* __restrict__ hv_g,
                                  const float* __restrict__ mpart,
                                  const float* __restrict__ nu1_w,
                                  const float* __restrict__ nu1_b,
                                  const float* __restrict__ nu2_w,
                                  const float* __restrict__ nu2_b,
                                  const float* __restrict__ nu3_w,
                                  const float* __restrict__ nu3_b,
                                  const float* __restrict__ dx_w,
                                  const float* __restrict__ dx_b,
                                  float* __restrict__ dxt) {
    __shared__ float nuin[2 * HH];
    __shared__ float ta[HH];
    __shared__ float tb[HH];
    const int row = blockIdx.x;
    const int f = threadIdx.x;
    const float hv = hv_g[row * HH + f];
    const float mv = mpart[row * HH + f] +
                     mpart[(size_t)BATCH * NN * HH + row * HH + f];
    nuin[f] = hv;
    nuin[HH + f] = mv;
    __syncthreads();
    float s = nu1_b[f];
    for (int k = 0; k < 2 * HH; ++k) s += nuin[k] * nu1_w[k * HH + f];
    ta[f] = silu_f(s);
    __syncthreads();
    s = nu2_b[f];
    for (int k = 0; k < HH; ++k) s += ta[k] * nu2_w[k * HH + f];
    tb[f] = silu_f(s);
    __syncthreads();
    s = nu3_b[f];
    for (int k = 0; k < HH; ++k) s += tb[k] * nu3_w[k * HH + f];
    const float hnew = hv + s;
    __syncthreads();
    nuin[f] = hnew;
    __syncthreads();
    if (f < 3) {
        float d = dx_b[f];
        for (int k = 0; k < HH; ++k) d += nuin[k] * dx_w[k * 3 + f];
        dxt[row * 3 + f] = tanhf(d);
    }
}

// ---------------------------------------------------------------------------
// Kernel 4: per-batch mean subtraction + softplus(bf_raw) scale.
// ---------------------------------------------------------------------------
__global__ void bf_final_kernel(const float* __restrict__ dxt,
                                const float* __restrict__ bf,
                                float* __restrict__ out) {
    __shared__ float mean[3];
    const int b = blockIdx.x;
    const int tid = threadIdx.x;
    if (tid < 3) {
        float s = 0.f;
        for (int n = 0; n < NN; ++n) s += dxt[(b * NN + n) * 3 + tid];
        mean[tid] = s * (1.0f / (float)NN);
    }
    __syncthreads();
    const float spv = log1pf(__expf(bf[0]));
    for (int idx = tid; idx < NN * 3; idx += 128) {
        const int c = idx % 3;
        out[b * NN * 3 + idx] = (dxt[b * NN * 3 + idx] - mean[c]) * spv;
    }
}

// ---------------------------------------------------------------------------
extern "C" void kernel_launch(void* const* d_in, const int* in_sizes, int n_in,
                              void* d_out, int out_size, void* d_ws, size_t ws_size,
                              hipStream_t stream) {
    const float* x     = (const float*)d_in[0];
    const int*   spin  = (const int*)d_in[1];
    const float* ne_w  = (const float*)d_in[2];
    const float* ne_b  = (const float*)d_in[3];
    const float* ee1_w = (const float*)d_in[4];
    const float* ee1_b = (const float*)d_in[5];
    const float* ee2_w = (const float*)d_in[6];
    const float* ee2_b = (const float*)d_in[7];
    const float* rve_w = (const float*)d_in[8];
    const float* rev_w = (const float*)d_in[9];
    const float* eu1_w = (const float*)d_in[10];
    const float* eu1_b = (const float*)d_in[11];
    const float* eu2_w = (const float*)d_in[12];
    const float* eu2_b = (const float*)d_in[13];
    const float* nu1_w = (const float*)d_in[14];
    const float* nu1_b = (const float*)d_in[15];
    const float* nu2_w = (const float*)d_in[16];
    const float* nu2_b = (const float*)d_in[17];
    const float* nu3_w = (const float*)d_in[18];
    const float* nu3_b = (const float*)d_in[19];
    const float* dx_w  = (const float*)d_in[20];
    const float* dx_b  = (const float*)d_in[21];
    const float* bf    = (const float*)d_in[22];
    float* out = (float*)d_out;

    char* ws = (char*)d_ws;
    size_t off = 0;
    auto take = [&](size_t bytes) -> char* {
        char* p = ws + off;
        off += (bytes + 255) & ~(size_t)255;
        return p;
    };
    _Float16* wE   = (_Float16*)take((size_t)WSWZ_H * 2);
    _Float16* wU   = (_Float16*)take((size_t)WSWZ_H * 2);
    _Float16* wM   = (_Float16*)take((size_t)WSWZ_H * 2);
    float*    bmsg = (float*)take(HH * 4);
    float*    hv   = (float*)take((size_t)BATCH * NN * HH * 4);
    float*    prei = (float*)take((size_t)BATCH * NN * HH * 4);
    float*    vjc  = (float*)take((size_t)BATCH * NN * HH * 4);
    float*    mprt = (float*)take((size_t)2 * BATCH * NN * HH * 4);
    float*    dxt  = (float*)take((size_t)BATCH * NN * 3 * 4);

    bf_prep_kernel<<<128, 128, 0, stream>>>(ee2_w, eu1_w, eu2_w, rev_w, eu2_b,
                                            wE, wU, wM, bmsg);
    bf_nodepre_kernel<<<BATCH * NN, 128, 0, stream>>>(x, spin, ne_w, ne_b,
                                                      rve_w, eu1_w, eu1_b,
                                                      hv, prei, vjc);
    // Dynamic LDS: 5216 floats + (3*WSWZ_H + 2*256*SA) halfs = 307,584 bytes.
    constexpr size_t SMEM_BYTES =
        (size_t)(640 + 128 + 128 + 128 + 2048 + 2048 + 48 + 48) * 4 +
        (size_t)(3 * WSWZ_H + 2 * 256 * SA) * 2;
    bf_edge_kernel<<<BATCH * 8 * 2, 256, SMEM_BYTES, stream>>>(
        x, wE, wU, wM, ee1_w, ee1_b, ee2_b, bmsg, prei, vjc, mprt);
    bf_nodeupd_kernel<<<BATCH * NN, 128, 0, stream>>>(hv, mprt, nu1_w, nu1_b,
                                                      nu2_w, nu2_b, nu3_w, nu3_b,
                                                      dx_w, dx_b, dxt);
    bf_final_kernel<<<BATCH, 128, 0, stream>>>(dxt, bf, out);
}
